// GraphAttentionLayer_35871566856642
// MI455X (gfx1250) — matrix operand order repaired and checked by reading, not compiled, not run on hardware
//
#include <hip/hip_runtime.h>

typedef __attribute__((ext_vector_type(16))) __bf16 v16bf;
typedef __attribute__((ext_vector_type(8)))  __bf16 v8bf;
typedef __attribute__((ext_vector_type(4)))  __bf16 v4bf;
typedef __attribute__((ext_vector_type(8)))  float  v8f;

#define GAT_B 8
#define GAT_N 2048
#define GAT_D 512
#define ALPHA 0.1f
#define NEG_INF_V -1e9f
#define LDP 2056         // padded LDS row stride (elements) -> conflict-free A-tile reads
#define MT 32            // rows per workgroup in the attention kernel

// ---------------------------------------------------------------- convert
__global__ void cvt_f32_bf16(const float* __restrict__ src, __bf16* __restrict__ dst, int n) {
    int i = blockIdx.x * blockDim.x + threadIdx.x;
    if (i < n) dst[i] = (__bf16)src[i];
}

// ------------------------------------------------- hidden = feat @ W (bf16 WMMA)
// grid.x = (B*N)/32 row tiles, block = 256 (8 waves); wave w owns columns [64w, 64w+64)
__global__ __launch_bounds__(256, 1)
void gemm_hidden(const __bf16* __restrict__ A,  // [B*N, 512] row major
                 const __bf16* __restrict__ W,  // [512, 512] row major
                 __bf16* __restrict__ H) {      // [B*N, 512]
    const int lane = threadIdx.x & 31;
    const int wave = threadIdx.x >> 5;
    const int row0 = blockIdx.x * 32;
    const int n0   = wave * 64;

    const int m  = lane & 15;
    const int k0 = (lane >> 4) << 3;   // 0 or 8: split-K halves of the 16-bit A layout

    const v8f vzero = {0.f,0.f,0.f,0.f,0.f,0.f,0.f,0.f};
    v8f acc[2][4];
#pragma unroll
    for (int h = 0; h < 2; ++h)
#pragma unroll
        for (int t = 0; t < 4; ++t) acc[h][t] = vzero;

    const __bf16* ap0 = A + (size_t)(row0 + m) * GAT_D + k0;        // m-half 0
    const __bf16* ap1 = ap0 + (size_t)16 * GAT_D;                   // m-half 1
    const __bf16* wp  = W + (size_t)k0 * GAT_D + n0 + m;            // lane = N column, K offset k0

    for (int kb = 0; kb < GAT_D; kb += 32) {
        // A tiles 16x32: lane<16 rows hold K [0..7]+[16..23], lanes>=16 hold [8..15]+[24..31]
        union { v16bf v; struct { v8bf lo, hi; } s; } u0, u1;
        u0.s.lo = *(const v8bf*)(ap0 + kb);
        u0.s.hi = *(const v8bf*)(ap0 + kb + 16);
        u1.s.lo = *(const v8bf*)(ap1 + kb);
        u1.s.hi = *(const v8bf*)(ap1 + kb + 16);
        const v16bf a0 = u0.v, a1 = u1.v;
#pragma unroll
        for (int t = 0; t < 4; ++t) {
            // B tile 32x16: lane = N column, K in table order (k0+[0..7], k0+16+[0..7]); reused by both halves
            v16bf bt;
            for (int s = 0; s < 16; ++s) bt[s] = wp[(size_t)(kb + (s < 8 ? s : s + 8)) * GAT_D + 16 * t];
            acc[0][t] = __builtin_amdgcn_wmma_f32_16x16x32_bf16(
                false, a0, false, bt, (short)0, acc[0][t], false, false);
            acc[1][t] = __builtin_amdgcn_wmma_f32_16x16x32_bf16(
                false, a1, false, bt, (short)0, acc[1][t], false, false);
        }
    }

    // C layout: VGPR r, lanes 0-15 -> (M=r, N=lane); lanes 16-31 -> (M=8+r, N=lane-16)
#pragma unroll
    for (int h = 0; h < 2; ++h)
#pragma unroll
        for (int t = 0; t < 4; ++t)
#pragma unroll
            for (int r = 0; r < 8; ++r) {
                const int mm = h * 16 + r + ((lane >> 4) << 3);
                const int nn = n0 + 16 * t + (lane & 15);
                H[(size_t)(row0 + mm) * GAT_D + nn] = (__bf16)acc[h][t][r];
            }
}

// ------------------------------------------- s1 = H@a1, s2 = H@a2 (one wave/row)
__global__ void s12_kernel(const __bf16* __restrict__ H, const float* __restrict__ a1,
                           const float* __restrict__ a2, float* __restrict__ s1,
                           float* __restrict__ s2, int rows) {
    const int row  = blockIdx.x * (blockDim.x >> 5) + (threadIdx.x >> 5);
    const int lane = threadIdx.x & 31;
    if (row >= rows) return;
    const __bf16* h = H + (size_t)row * GAT_D;
    float d1 = 0.f, d2 = 0.f;
    for (int k = lane; k < GAT_D; k += 32) {
        const float hv = (float)h[k];
        d1 += hv * a1[k];
        d2 += hv * a2[k];
    }
#pragma unroll
    for (int off = 16; off; off >>= 1) {
        d1 += __shfl_down(d1, off, 32);
        d2 += __shfl_down(d2, off, 32);
    }
    if (lane == 0) { s1[row] = d1; s2[row] = d2; }
}

// --------------------------------- fused: scores -> softmax -> attn@hidden -> ELU
// grid.x = B * (N/32); block = 256 (8 waves); 32 output rows per workgroup
__global__ __launch_bounds__(256, 1)
void gat_attn(const int* __restrict__ graph,   // [B,N,N] int32 (0/1)
              const float* __restrict__ s1,    // [B*N]
              const float* __restrict__ s2,    // [B*N]
              const __bf16* __restrict__ H,    // [B,N,512] bf16
              float* __restrict__ out) {       // [B,N,512] f32
    __shared__ __align__(32) __bf16 P[MT * LDP];  // unnormalized probs (bf16), ~128.5 KB
    __shared__ float wred[8 * MT];
    __shared__ float rowMax[MT];
    __shared__ float rowSum[MT];

    const int tid  = threadIdx.x;
    const int lane = tid & 31;
    const int wave = tid >> 5;
    const int b    = blockIdx.x >> 6;           // N/32 = 64 row tiles per batch
    const int row0 = (blockIdx.x & 63) << 5;
    const long long base = (long long)b * GAT_N;
    const float* s2b = s2 + base;

    // ---- Phase A: masked leaky-relu scores into LDS (vectorized), track row max
    for (int i = 0; i < MT; ++i) {
        const int*  grow = graph + (base + row0 + i) * (long long)GAT_N;
        const float si = s1[base + row0 + i];
        float lmax = -3.4e38f;
#pragma unroll
        for (int it = 0; it < 2; ++it) {
            const int j = (tid << 2) + (it << 10);           // 4 consecutive cols/thread
            const int4   g = *(const int4*)(grow + j);
            const float4 t2 = *(const float4*)(s2b + j);
            float e0 = si + t2.x; e0 = (e0 > 0.f) ? e0 : ALPHA * e0; if (g.x == 0) e0 = NEG_INF_V;
            float e1 = si + t2.y; e1 = (e1 > 0.f) ? e1 : ALPHA * e1; if (g.y == 0) e1 = NEG_INF_V;
            float e2 = si + t2.z; e2 = (e2 > 0.f) ? e2 : ALPHA * e2; if (g.z == 0) e2 = NEG_INF_V;
            float e3 = si + t2.w; e3 = (e3 > 0.f) ? e3 : ALPHA * e3; if (g.w == 0) e3 = NEG_INF_V;
            v4bf pv = {(__bf16)e0, (__bf16)e1, (__bf16)e2, (__bf16)e3};
            *(v4bf*)&P[i * LDP + j] = pv;
            lmax = fmaxf(fmaxf(fmaxf(e0, e1), fmaxf(e2, e3)), lmax);
        }
#pragma unroll
        for (int off = 16; off; off >>= 1) lmax = fmaxf(lmax, __shfl_down(lmax, off, 32));
        if (lane == 0) wred[wave * MT + i] = lmax;
    }
    __syncthreads();
    if (tid < MT) {
        float mx = wred[tid];
#pragma unroll
        for (int w = 1; w < 8; ++w) mx = fmaxf(mx, wred[w * MT + tid]);
        rowMax[tid] = mx;
    }
    __syncthreads();

    // ---- Phase B: exponentiate in place (8 elems/thread/row), accumulate row sums
    for (int i = 0; i < MT; ++i) {
        const float mx = rowMax[i];
        const int j = tid << 3;
        v8bf pv = *(v8bf*)&P[i * LDP + j];
        float lsum = 0.f;
        v8bf pw;
#pragma unroll
        for (int k = 0; k < 8; ++k) {
            const float p = __expf((float)pv[k] - mx);
            pw[k] = (__bf16)p;
            lsum += p;
        }
        *(v8bf*)&P[i * LDP + j] = pw;
#pragma unroll
        for (int off = 16; off; off >>= 1) lsum += __shfl_down(lsum, off, 32);
        if (lane == 0) wred[wave * MT + i] = lsum;
    }
    __syncthreads();
    if (tid < MT) {
        float s = 0.f;
#pragma unroll
        for (int w = 0; w < 8; ++w) s += wred[w * MT + tid];
        rowSum[tid] = s;
    }
    __syncthreads();

    // ---- Phase C: out_block[32,512] = P[32,2048] @ H_b[2048,512], WMMA bf16
    const int n0 = wave << 6;                       // 64 columns per wave
    const v8f vzero = {0.f,0.f,0.f,0.f,0.f,0.f,0.f,0.f};
    v8f acc[2][4];
#pragma unroll
    for (int h = 0; h < 2; ++h)
#pragma unroll
        for (int t = 0; t < 4; ++t) acc[h][t] = vzero;

    const int m  = lane & 15;
    const int k0 = (lane >> 4) << 3;
    const __bf16* hp = H + (size_t)b * GAT_N * GAT_D + (size_t)k0 * GAT_D + n0 + m;

    for (int kb = 0; kb < GAT_N; kb += 32) {
        union { v16bf v; struct { v8bf lo, hi; } s; } u0, u1;
        u0.s.lo = *(const v8bf*)&P[m * LDP + kb + k0];              // ds_load_b128
        u0.s.hi = *(const v8bf*)&P[m * LDP + kb + k0 + 16];
        u1.s.lo = *(const v8bf*)&P[(16 + m) * LDP + kb + k0];
        u1.s.hi = *(const v8bf*)&P[(16 + m) * LDP + kb + k0 + 16];
        const v16bf a0 = u0.v, a1 = u1.v;
#pragma unroll
        for (int t = 0; t < 4; ++t) {
            v16bf bt;
            for (int s = 0; s < 16; ++s) bt[s] = hp[(size_t)(kb + (s < 8 ? s : s + 8)) * GAT_D + 16 * t];
            acc[0][t] = __builtin_amdgcn_wmma_f32_16x16x32_bf16(
                false, a0, false, bt, (short)0, acc[0][t], false, false);
            acc[1][t] = __builtin_amdgcn_wmma_f32_16x16x32_bf16(
                false, a1, false, bt, (short)0, acc[1][t], false, false);
        }
    }

    // scale by 1/rowsum (softmax), ELU, store f32
#pragma unroll
    for (int h = 0; h < 2; ++h)
#pragma unroll
        for (int t = 0; t < 4; ++t)
#pragma unroll
            for (int r = 0; r < 8; ++r) {
                const int mm = h * 16 + r + ((lane >> 4) << 3);
                float v = acc[h][t][r] / rowSum[mm];
                v = (v > 0.f) ? v : (__expf(v) - 1.f);
                out[((size_t)(base + row0 + mm)) * GAT_D + n0 + 16 * t + (lane & 15)] = v;
            }
}

// ----------------------------------------------------------------- launcher
extern "C" void kernel_launch(void* const* d_in, const int* in_sizes, int n_in,
                              void* d_out, int out_size, void* d_ws, size_t ws_size,
                              hipStream_t stream) {
    const float* features = (const float*)d_in[0];      // [8,2048,512] f32
    const int*   graph    = (const int*)d_in[1];        // [8,2048,2048] int32
    const float* w        = (const float*)d_in[2];      // [512,512] f32
    const float* a1       = (const float*)d_in[3];      // [512] f32
    const float* a2       = (const float*)d_in[4];      // [512] f32
    float* out            = (float*)d_out;

    const int ROWS = GAT_B * GAT_N;                     // 16384
    const size_t featElems = (size_t)ROWS * GAT_D;      // 8388608
    const size_t wElems    = (size_t)GAT_D * GAT_D;     // 262144

    char* ws = (char*)d_ws;
    __bf16* feat_bf = (__bf16*)ws;                                  // 16 MB
    __bf16* w_bf    = (__bf16*)(ws + featElems * 2);                // 512 KB
    __bf16* hid_bf  = (__bf16*)(ws + featElems * 2 + wElems * 2);   // 16 MB
    float*  s1      = (float*)(ws + featElems * 4 + wElems * 2);
    float*  s2      = s1 + ROWS;

    cvt_f32_bf16<<<(int)((featElems + 255) / 256), 256, 0, stream>>>(features, feat_bf, (int)featElems);
    cvt_f32_bf16<<<(int)((wElems + 255) / 256), 256, 0, stream>>>(w, w_bf, (int)wElems);

    gemm_hidden<<<ROWS / 32, 256, 0, stream>>>(feat_bf, w_bf, hid_bf);

    s12_kernel<<<ROWS / 8, 256, 0, stream>>>(hid_bf, a1, a2, s1, s2, ROWS);

    gat_attn<<<GAT_B * (GAT_N / 32), 256, 0, stream>>>(graph, s1, s2, hid_bf, out);
}
